// RandLANet_78623671320909
// MI455X (gfx1250) — compile-verified, matrix operand order checked by reading, not run on hardware
//
#include <hip/hip_runtime.h>
#include <hip/hip_bf16.h>

// ---------------------------------------------------------------------------
// RandLA-Net forward for MI455X (gfx1250). All dense layers go through
// v_wmma_f32_16x16x32_f16 with fused bias / BN / activation epilogue.
// The GEMM K-loop is software-pipelined: next tile's global_load_b128s are
// issued while the current tile's ds_load + v_wmma execute, so the loadcnt
// wait lands one iteration later and overlaps the matrix pipe.
// ---------------------------------------------------------------------------

typedef __attribute__((ext_vector_type(16))) _Float16 v16h;
typedef __attribute__((ext_vector_type(8)))  float    v8f;
typedef _Float16 h4   __attribute__((ext_vector_type(4)));             // 8B LDS store
typedef float    f4u  __attribute__((ext_vector_type(4), aligned(4))); // dword-aligned b128 load

#define BN_SCALE_F 0.9999995000003750f   // 1/sqrt(1+1e-6)

// Load one 4-wide f32 segment, convert to f16. interior => single b128 load;
// boundary => index-clamped scalar loads + selects (no exec divergence).
static __device__ __forceinline__ h4 stage_seg(const float* __restrict__ Mtx, int ld,
                                               int nrows, int ncols, int r, int c,
                                               bool interior)
{
    h4 hv;
    if (interior) {
        f4u v = *(const f4u*)(Mtx + (size_t)r * ld + c);
        hv[0] = (_Float16)v[0]; hv[1] = (_Float16)v[1];
        hv[2] = (_Float16)v[2]; hv[3] = (_Float16)v[3];
    } else {
        int rc = r < nrows - 1 ? r : nrows - 1;
        const float* base = Mtx + (size_t)rc * ld;
        #pragma unroll
        for (int i = 0; i < 4; ++i) {
            int gc = c + i; int gcc = gc < ncols - 1 ? gc : ncols - 1;
            float v = base[gcc];                         // always in-bounds
            hv[i] = (_Float16)((r < nrows && gc < ncols) ? v : 0.0f);
        }
    }
    return hv;
}

// ============================ WMMA GEMM ====================================
// C[m, coff+n] = act( (A(M,K) @ W(N,K)^T + bias)[*bn] ), row-major, strided.
// Block: 256 threads = 8 waves as 2(M) x 4(N) 16x16 WMMA tiles -> 32x64 tile.
__global__ __launch_bounds__(256)
void gemm_wmma_kernel(const float* __restrict__ A, int lda,
                      const float* __restrict__ W,          // (N,K) row-major
                      const float* __restrict__ bias,       // nullable
                      const float* __restrict__ bng,        // nullable
                      const float* __restrict__ bnb,        // nullable
                      float* __restrict__ C, int ldc, int coff,
                      int M, int N, int K,
                      int act, float slope)
{
    __shared__ _Float16 As[32][40];   // 40-half row pitch: 8B aligned segs, bank-skewed
    __shared__ _Float16 Ws[64][40];

    const int tid   = threadIdx.x;
    const int lane  = tid & 31;
    const int wv    = tid >> 5;       // 0..7
    const int wr    = wv >> 2;        // 0..1 (M tile)
    const int wc    = wv & 3;         // 0..3 (N tile)
    const int tileM = blockIdx.y * 32;
    const int tileN = blockIdx.x * 64;

    // Per-thread staging coords: one 4-wide segment of the 32x32 A tile,
    // two 4-wide segments of the 64x32 W tile.
    const int ar = tid >> 3;          // 0..31
    const int ac = (tid & 7) * 4;     // 0,4,..,28
    const int w0r = tid >> 3,        w0c = (tid & 7) * 4;
    const int w1r = (tid + 256) >> 3, w1c = w0c;

    const bool aInt = (tileM + 32 <= M);
    const bool wInt = (tileN + 64 <= N);

    v8f acc = {};

    // -------- pipeline prologue: stage K-tile 0 into registers --------
    bool kInt = (32 <= K);
    h4 aReg  = stage_seg(A, lda, M, K, tileM + ar,  0 + ac,  aInt && kInt);
    h4 wReg0 = stage_seg(W, K,   N, K, tileN + w0r, 0 + w0c, wInt && kInt);
    h4 wReg1 = stage_seg(W, K,   N, K, tileN + w1r, 0 + w1c, wInt && kInt);

    for (int k0 = 0; k0 < K; k0 += 32) {
        // commit current register tile to LDS
        *(h4*)(&As[ar][ac])   = aReg;
        *(h4*)(&Ws[w0r][w0c]) = wReg0;
        *(h4*)(&Ws[w1r][w1c]) = wReg1;

        // issue next tile's global loads; their wait lands at next commit,
        // overlapping this iteration's ds_load + wmma + barriers
        const int kn = k0 + 32;
        if (kn < K) {
            bool knInt = (kn + 32 <= K);
            aReg  = stage_seg(A, lda, M, K, tileM + ar,  kn + ac,  aInt && knInt);
            wReg0 = stage_seg(W, K,   N, K, tileN + w0r, kn + w0c, wInt && knInt);
            wReg1 = stage_seg(W, K,   N, K, tileN + w1r, kn + w1c, wInt && knInt);
            if (kn + 32 < K) {  // pull the tile after that toward L2/L0
                int pr = tileM + ar; if (pr >= M) pr = M - 1;
                __builtin_prefetch(A + (size_t)pr * lda + kn + 32, 0, 0);
                int pw = tileN + (tid >> 2); if (pw >= N) pw = N - 1;
                __builtin_prefetch(W + (size_t)pw * K + kn + 32, 0, 0);
            }
        }
        __syncthreads();

        // Gather WMMA fragments per ISA 7.12.2 (16-bit A 16x32 layout; B mirrored).
        v16h af, bf;
        const int half = (lane >> 4) << 3;   // lanes 16-31 -> K+8
        const int rsel = lane & 15;          // A row / B column within tile
        #pragma unroll
        for (int j = 0; j < 16; ++j) {
            int vv = j >> 1, ws = j & 1;
            int kk = ((vv & 4) << 2) + half + ((vv & 3) << 1) + ws;
            af[j] = As[wr * 16 + rsel][kk];
            bf[j] = Ws[wc * 16 + rsel][kk];
        }
        acc = __builtin_amdgcn_wmma_f32_16x16x32_f16(
                  false, af, false, bf, (short)0, acc, false, false);
        __syncthreads();
    }

    // Epilogue: C/D layout -> VGPR r: M=r (lanes 0-15) / r+8 (lanes 16-31), N=lane&15.
    const int nloc = tileN + wc * 16 + (lane & 15);
    #pragma unroll
    for (int r = 0; r < 8; ++r) {
        int mloc = tileM + wr * 16 + r + ((lane >> 4) << 3);
        if (mloc < M && nloc < N) {
            float v = acc[r];
            if (bias) v += bias[nloc];
            if (bng)  v = v * (bng[nloc] * BN_SCALE_F) + bnb[nloc];
            if (act == 1)      v = v > 0.0f ? v : 0.0f;
            else if (act == 2) v = v > 0.0f ? v : v * slope;
            C[(size_t)mloc * ldc + coff + nloc] = v;
        }
    }
}

// ============================== kNN (K=16) =================================
#define KNN_K 16
__global__ __launch_bounds__(128)
void knn_kernel(const float* __restrict__ coords, int n,
                int* __restrict__ idx_out, float* __restrict__ dist_out)
{
    __shared__ float sx[128], sy[128], sz[128];
    const int q = blockIdx.x * 128 + threadIdx.x;
    const bool active = q < n;
    float qx = 0.f, qy = 0.f, qz = 0.f;
    if (active) { qx = coords[q*3+0]; qy = coords[q*3+1]; qz = coords[q*3+2]; }

    float bd[KNN_K]; int bi[KNN_K];
    #pragma unroll
    for (int i = 0; i < KNN_K; ++i) { bd[i] = 3.4e38f; bi[i] = 0; }

    for (int t0 = 0; t0 < n; t0 += 128) {
        __syncthreads();
        int r = t0 + threadIdx.x;
        if (r < n) {
            sx[threadIdx.x] = coords[r*3+0];
            sy[threadIdx.x] = coords[r*3+1];
            sz[threadIdx.x] = coords[r*3+2];
        }
        __syncthreads();
        int lim = n - t0; if (lim > 128) lim = 128;
        if (active) {
            for (int j = 0; j < lim; ++j) {
                float dx = qx - sx[j], dy = qy - sy[j], dz = qz - sz[j];
                float d2 = dx*dx + dy*dy + dz*dz;
                if (d2 < bd[KNN_K-1]) {
                    bd[KNN_K-1] = d2; bi[KNN_K-1] = t0 + j;
                    #pragma unroll
                    for (int i = KNN_K-1; i > 0; --i) {
                        if (bd[i] < bd[i-1]) {
                            float td = bd[i]; bd[i] = bd[i-1]; bd[i-1] = td;
                            int   ti = bi[i]; bi[i] = bi[i-1]; bi[i-1] = ti;
                        }
                    }
                }
            }
        }
    }
    if (active) {
        #pragma unroll
        for (int i = 0; i < KNN_K; ++i) {
            idx_out[q*KNN_K + i]  = bi[i];
            dist_out[q*KNN_K + i] = sqrtf(bd[i] > 0.f ? bd[i] : 0.f);
        }
    }
}

// Decoder upsample: nearest among coords[:n_ref] for each of coords[:n_q].
__global__ void knn1_kernel(const float* __restrict__ coords, int n_ref, int n_q,
                            int* __restrict__ idx_out)
{
    int q = blockIdx.x * 256 + threadIdx.x;
    if (q >= n_q) return;
    float qx = coords[q*3+0], qy = coords[q*3+1], qz = coords[q*3+2];
    float best = 3.4e38f; int bidx = 0;
    for (int r = 0; r < n_ref; ++r) {
        float dx = qx - coords[r*3+0], dy = qy - coords[r*3+1], dz = qz - coords[r*3+2];
        float d2 = dx*dx + dy*dy + dz*dz;
        if (d2 < best) { best = d2; bidx = r; }
    }
    idx_out[q] = bidx;
}

// =========================== glue kernels ==================================
// geo = [ctr(3), nbr(3), ctr-nbr(3), dist(1)] -> (n*16, 10)
__global__ void build_geo_kernel(const float* __restrict__ coords,
                                 const int* __restrict__ idx,
                                 const float* __restrict__ dist,
                                 float* __restrict__ geo, int n)
{
    int t = blockIdx.x * 256 + threadIdx.x;
    if (t >= n * 16) return;
    int q = t >> 4;
    int r = idx[t];
    float cx = coords[q*3+0], cy = coords[q*3+1], cz = coords[q*3+2];
    float nx = coords[r*3+0], ny = coords[r*3+1], nz = coords[r*3+2];
    float* g = geo + (size_t)t * 10;
    g[0]=cx; g[1]=cy; g[2]=cz; g[3]=nx; g[4]=ny; g[5]=nz;
    g[6]=cx-nx; g[7]=cy-ny; g[8]=cz-nz; g[9]=dist[t];
}

// Generic row gather/copy: dst[i, off+c] = src[map?map[i]:i, c]
__global__ void gather_rows_kernel(float* __restrict__ dst, int dst_ld, int dst_off,
                                   const float* __restrict__ src, int src_ld, int ncols,
                                   const int* __restrict__ map, int nrows)
{
    long long t = (long long)blockIdx.x * 256 + threadIdx.x;
    if (t >= (long long)nrows * ncols) return;
    int i = (int)(t / ncols), c = (int)(t % ncols);
    int r = map ? map[i] : i;
    dst[(size_t)i * dst_ld + dst_off + c] = src[(size_t)r * src_ld + c];
}

// softmax over K=16 (axis 2) + weighted sum: F[q,c] = sum_k softmax_k(S)*X
__global__ void attpool_softmax_kernel(const float* __restrict__ S,
                                       const float* __restrict__ X,
                                       float* __restrict__ F, int n, int C)
{
    long long t = (long long)blockIdx.x * 256 + threadIdx.x;
    if (t >= (long long)n * C) return;
    int q = (int)(t / C), ch = (int)(t % C);
    const float* s = S + (size_t)q * 16 * C + ch;
    const float* x = X + (size_t)q * 16 * C + ch;
    float m = -3.4e38f;
    #pragma unroll
    for (int k = 0; k < 16; ++k) { float v = s[(size_t)k * C]; m = v > m ? v : m; }
    float sum = 0.f, accv = 0.f;
    #pragma unroll
    for (int k = 0; k < 16; ++k) {
        float e = __expf(s[(size_t)k * C] - m);
        sum += e; accv += e * x[(size_t)k * C];
    }
    F[t] = accv / sum;
}

__global__ void add_leaky_kernel(const float* __restrict__ a, const float* __restrict__ b,
                                 float* __restrict__ o, long long n, float slope)
{
    long long t = (long long)blockIdx.x * 256 + threadIdx.x;
    if (t >= n) return;
    float v = a[t] + b[t];
    o[t] = v > 0.f ? v : v * slope;
}

__global__ void inv_perm_kernel(const int* __restrict__ perm, int* __restrict__ inv, int n)
{
    int t = blockIdx.x * 256 + threadIdx.x;
    if (t < n) inv[perm[t]] = t;
}

__global__ void transpose_kernel(float* __restrict__ dst, const float* __restrict__ src,
                                 int rows, int cols)
{
    long long t = (long long)blockIdx.x * 256 + threadIdx.x;
    if (t >= (long long)rows * cols) return;
    int i = (int)(t / cols), c = (int)(t % cols);
    dst[(size_t)c * rows + i] = src[(size_t)i * cols + c];
}

__global__ void perm_to_float_kernel(const int* __restrict__ perm, float* __restrict__ out, int n)
{
    int t = blockIdx.x * 256 + threadIdx.x;
    if (t < n) out[t] = (float)perm[t];
}

// ============================ host orchestration ===========================
static inline int cdiv(int a, int b) { return (a + b - 1) / b; }

static void gemm(hipStream_t s, const float* A, int lda, const float* W,
                 const float* b, const float* g, const float* bt,
                 float* C, int ldc, int coff, int M, int N, int K,
                 int act, float slope)
{
    dim3 grid(cdiv(N, 64), cdiv(M, 32));
    gemm_wmma_kernel<<<grid, 256, 0, s>>>(A, lda, W, b, g, bt, C, ldc, coff,
                                          M, N, K, act, slope);
}

extern "C" void kernel_launch(void* const* d_in, const int* in_sizes, int n_in,
                              void* d_out, int out_size, void* d_ws, size_t ws_size,
                              hipStream_t stream)
{
    (void)in_sizes; (void)n_in; (void)out_size; (void)ws_size;
    const int N = 8192;

    // ---- inputs (JAX pytree order: dict keys sorted) ----
    const float* inp  = (const float*)d_in[0];   // (8192, 6)
    const float* nbri = (const float*)d_in[1];   // (8192, 343)
    const int*   perm = (const int*)d_in[2];     // (8192)
    auto P = [&](int i) -> const float* { return (const float*)d_in[i]; };
    // 3: bn_start.bt  4: bn_start.g
    // 5: conv3d.b  6: conv3d.w  7: conv_fc.b  8: conv_fc.w
    // 9..24: dec[0..3] each {b, bt, g, w}
    // 25 + 26*L: enc level L:
    //   +0..3 lse1{b,bt,g,w} +4..7 lse2 +8 mlp1.b +9 mlp1.w +10 mlp2.b +11 mlp2.w
    //   +12..15 pool1.mlp{b,bt,g,w} +16 pool1.score_w
    //   +17..20 pool2.mlp{b,bt,g,w} +21 pool2.score_w  +22..25 shortcut{b,bt,g,w}
    // 129..132 fc_end1{b,bt,g,w}  133..136 fc_end2  137 fc_out.b 138 fc_out.w
    // 139 fc_start.b 140 fc_start.w  141 mid.b 142 mid.w

    // ---- workspace bump allocator (ws poisoned; we write before reading) ----
    char* wsp = (char*)d_ws;
    auto alloc = [&](size_t bytes) -> char* {
        char* p = wsp; wsp += (bytes + 255) & ~(size_t)255; return p;
    };
    float* coords = (float*)alloc((size_t)N*3*4);
    int*   inv    = (int*)  alloc((size_t)N*4);
    int*   idx1   = (int*)  alloc((size_t)N*4);
    int*   idxK   = (int*)  alloc((size_t)N*16*4);
    float* distK  = (float*)alloc((size_t)N*16*4);
    float* xcat   = (float*)alloc((size_t)N*80*4);
    float* xperm  = (float*)alloc((size_t)N*80*4);
    float* f1     = (float*)alloc((size_t)N*256*4);
    float* geo    = (float*)alloc((size_t)N*16*10*4);
    float* lseb   = (float*)alloc((size_t)2097152*4);   // max n*16*d_out
    float* Sbuf   = (float*)alloc((size_t)2097152*4);
    float* fpool  = (float*)alloc((size_t)262144*4);
    float* xL     = (float*)alloc((size_t)262144*4);
    float* x2     = (float*)alloc((size_t)262144*4);
    float* x3     = (float*)alloc((size_t)262144*4);
    float* t1     = (float*)alloc((size_t)262144*4);
    float* t2     = (float*)alloc((size_t)262144*4);
    float* stk[4];
    stk[0] = (float*)alloc((size_t)8192*32*4);
    stk[1] = (float*)alloc((size_t)2048*128*4);
    stk[2] = (float*)alloc((size_t)512*256*4);
    stk[3] = (float*)alloc((size_t)128*512*4);
    float* lv     = (float*)alloc((size_t)32*512*4);
    float* dcat   = (float*)alloc((size_t)8192*64*4);
    float* douta  = (float*)alloc((size_t)262144*4);
    float* doutb  = (float*)alloc((size_t)262144*4);
    float* xun    = (float*)alloc((size_t)N*16*4);
    float* e1     = (float*)alloc((size_t)N*64*4);
    float* emb    = (float*)alloc((size_t)N*128*4);
    float* scbuf  = (float*)alloc((size_t)N*13*4);

    // ---- stem ----
    gemm(stream, inp, 6, P(140), P(139), P(4), P(3), xcat, 80, 0, N, 16, 6, 2, 0.2f);
    gemm(stream, nbri, 343, P(6), P(5), nullptr, nullptr, f1, 256, 0, N, 256, 343, 1, 0.f);
    gemm(stream, f1, 256, P(8), P(7), nullptr, nullptr, xcat, 80, 16, N, 64, 256, 1, 0.f);
    gather_rows_kernel<<<cdiv(N*3, 256), 256, 0, stream>>>(coords, 3, 0, inp, 6, 3, perm, N);
    gather_rows_kernel<<<cdiv(N*80, 256), 256, 0, stream>>>(xperm, 80, 0, xcat, 80, 80, perm, N);
    inv_perm_kernel<<<cdiv(N, 256), 256, 0, stream>>>(perm, inv, N);

    // ---- encoder ----
    const int nL[4]    = {8192, 2048, 512, 128};
    const int dinL[4]  = {80, 32, 128, 256};
    const int doutL[4] = {16, 64, 128, 256};
    const float* x_in = xperm;
    int din = 80;
    for (int L = 0; L < 4; ++L) {
        const int n = nL[L], d = doutL[L], h = d / 2, EB = 25 + 26 * L;
        din = dinL[L];
        knn_kernel<<<cdiv(n, 128), 128, 0, stream>>>(coords, n, idxK, distK);
        gemm(stream, x_in, din, P(EB+9), P(EB+8), nullptr, nullptr, xL, h, 0, n, h, din, 2, 0.2f);
        build_geo_kernel<<<cdiv(n*16, 256), 256, 0, stream>>>(coords, idxK, distK, geo, n);
        // lse1 -> att_pool1
        gemm(stream, geo, 10, P(EB+3), P(EB+0), P(EB+2), P(EB+1), lseb, d, 0, n*16, h, 10, 1, 0.f);
        gather_rows_kernel<<<cdiv(n*16*h, 256), 256, 0, stream>>>(lseb, d, h, xL, h, h, idxK, n*16);
        gemm(stream, lseb, d, P(EB+16), nullptr, nullptr, nullptr, Sbuf, d, 0, n*16, d, d, 0, 0.f);
        attpool_softmax_kernel<<<cdiv(n*d, 256), 256, 0, stream>>>(Sbuf, lseb, fpool, n, d);
        gemm(stream, fpool, d, P(EB+15), P(EB+12), P(EB+14), P(EB+13), x2, h, 0, n, h, d, 1, 0.f);
        // lse2 -> att_pool2
        gemm(stream, geo, 10, P(EB+7), P(EB+4), P(EB+6), P(EB+5), lseb, d, 0, n*16, h, 10, 1, 0.f);
        gather_rows_kernel<<<cdiv(n*16*h, 256), 256, 0, stream>>>(lseb, d, h, x2, h, h, idxK, n*16);
        gemm(stream, lseb, d, P(EB+21), nullptr, nullptr, nullptr, Sbuf, d, 0, n*16, d, d, 0, 0.f);
        attpool_softmax_kernel<<<cdiv(n*d, 256), 256, 0, stream>>>(Sbuf, lseb, fpool, n, d);
        gemm(stream, fpool, d, P(EB+20), P(EB+17), P(EB+19), P(EB+18), x3, d, 0, n, d, d, 1, 0.f);
        // out = leaky(lin(x3, mlp2) + bn(lin(feats, shortcut)), 0.01)
        gemm(stream, x3, d, P(EB+11), P(EB+10), nullptr, nullptr, t1, 2*d, 0, n, 2*d, d, 0, 0.f);
        gemm(stream, x_in, din, P(EB+25), P(EB+22), P(EB+24), P(EB+23), t2, 2*d, 0, n, 2*d, din, 0, 0.f);
        add_leaky_kernel<<<cdiv(n*2*d, 256), 256, 0, stream>>>(t1, t2, stk[L], (long long)n*2*d, 0.01f);
        x_in = stk[L];   // next level reads the contiguous row-prefix
    }

    // ---- mid ----
    gemm(stream, stk[3], 512, P(142), P(141), nullptr, nullptr, lv, 512, 0, 32, 512, 512, 1, 0.f);

    // ---- decoder ----
    const int decO[4]  = {256, 128, 32, 16};
    const int decK_[4] = {1024, 512, 256, 64};
    const int dbase[4] = {9, 13, 17, 21};
    const int nref[4]  = {32, 128, 512, 2048};
    const int nq[4]    = {128, 512, 2048, 8192};
    const float* xcur = lv;
    int cup = 512;
    float* dout = douta;
    for (int i = 0; i < 4; ++i) {
        knn1_kernel<<<cdiv(nq[i], 256), 256, 0, stream>>>(coords, nref[i], nq[i], idx1);
        gather_rows_kernel<<<cdiv(nq[i]*cup, 256), 256, 0, stream>>>(dcat, decK_[i], 0, xcur, cup, cup, idx1, nq[i]);
        gather_rows_kernel<<<cdiv(nq[i]*cup, 256), 256, 0, stream>>>(dcat, decK_[i], cup, stk[3-i], cup, cup, nullptr, nq[i]);
        const int B0 = dbase[i];
        gemm(stream, dcat, decK_[i], P(B0+3), P(B0+0), P(B0+2), P(B0+1),
             dout, decO[i], 0, nq[i], decO[i], decK_[i], 1, 0.f);
        xcur = dout; cup = decO[i];
        dout = (dout == douta) ? doutb : douta;
    }

    // ---- un-permute + head ----
    gather_rows_kernel<<<cdiv(N*16, 256), 256, 0, stream>>>(xun, 16, 0, xcur, 16, 16, inv, N);
    gemm(stream, xun, 16, P(132), P(129), P(131), P(130), e1, 64, 0, N, 64, 16, 1, 0.f);
    gemm(stream, e1, 64, P(136), P(133), P(135), P(134), emb, 128, 0, N, 128, 64, 1, 0.f);
    gemm(stream, emb, 128, P(138), P(137), nullptr, nullptr, scbuf, 13, 0, N, 13, 128, 0, 0.f);

    // ---- outputs: st(13,8192), sdf(13,8192), emb(128,8192), lv(512,32), perm(8192) ----
    float* out = (float*)d_out;
    transpose_kernel<<<cdiv(N*13, 256), 256, 0, stream>>>(out, scbuf, N, 13);
    transpose_kernel<<<cdiv(N*13, 256), 256, 0, stream>>>(out + 106496, scbuf, N, 13);
    transpose_kernel<<<cdiv(N*128, 256), 256, 0, stream>>>(out + 212992, emb, N, 128);
    transpose_kernel<<<cdiv(32*512, 256), 256, 0, stream>>>(out + 1261568, lv, 32, 512);
    perm_to_float_kernel<<<cdiv(N, 256), 256, 0, stream>>>(perm, out + 1277952, N);
}
